// HSTUBlock_41987600285923
// MI455X (gfx1250) — matrix operand order repaired and checked by reading, not compiled, not run on hardware
//
#include <hip/hip_runtime.h>
#include <hip/hip_bf16.h>

// ---------------------------------------------------------------------------
// HSTU block for MI455X (gfx1250), wave32 + WMMA bf16 (v_wmma_f32_16x16x32_bf16)
// + async global->LDS staging (GLOBAL_LOAD_ASYNC_TO_LDS_B128, ASYNCcnt)
// ---------------------------------------------------------------------------
#define BATCH 8
#define SEQ   2048
#define DIM   1024
#define HEADS 8
#define EPROJ 4096            // 2*H*DV + 2*H*DQ
#define NBUCK 128

typedef __bf16 bf16;
typedef __attribute__((ext_vector_type(16))) __bf16 v16bf;
typedef __attribute__((ext_vector_type(8)))  float  v8f;
typedef __attribute__((address_space(3))) bf16 as3_bf16;

__device__ __forceinline__ int imin(int a, int b) { return a < b ? a : b; }
__device__ __forceinline__ int imax(int a, int b) { return a > b ? a : b; }
__device__ __forceinline__ float silu_f(float x) { return x / (1.0f + __expf(-x)); }

__device__ __forceinline__ v8f wmma_bf16(v16bf a, v16bf b, v8f c) {
  return __builtin_amdgcn_wmma_f32_16x16x32_bf16(false, a, false, b, (short)0, c,
                                                 false, false);
}

// Per-lane async DMA: 16B global -> LDS, tracked by ASYNCcnt. No VGPR staging.
__device__ __forceinline__ void async_copy_b128(bf16* dst_lds, const bf16* src) {
  unsigned l = (unsigned)(unsigned long long)(as3_bf16*)dst_lds;
  asm volatile("global_load_async_to_lds_b128 %0, %1, off"
               :: "v"(l), "v"(src) : "memory");
}
__device__ __forceinline__ void wait_async0() {
  asm volatile("s_wait_asynccnt 0x0" ::: "memory");
}

// A fragment (16x32 bf16) from row-major LDS tile, stride in elements.
__device__ __forceinline__ v16bf load_fragA(const bf16* base, int stride,
                                            int row, int k0, int lane) {
  int half = (lane >> 4) & 1;
  int m    = lane & 15;
  const bf16* p = base + (size_t)(row + m) * stride + k0;
  union { v16bf v; uint4 q[2]; } f;
  f.q[0] = *reinterpret_cast<const uint4*>(p + half * 8);
  f.q[1] = *reinterpret_cast<const uint4*>(p + 16 + half * 8);
  return f.v;
}

// B fragment (32x16 bf16) from LDS tile stored TRANSPOSED: baseT[n][k].
__device__ __forceinline__ v16bf load_fragB(const bf16* baseT, int stride,
                                            int col, int k0, int lane) {
  int half = (lane >> 4) & 1;
  int n    = lane & 15;
  const bf16* p = baseT + (size_t)(col + n) * stride + k0 + half * 8;
  union { v16bf v; uint4 q[2]; } f;
  f.q[0] = *reinterpret_cast<const uint4*>(p);
  f.q[1] = *reinterpret_cast<const uint4*>(p + 16);
  return f.v;
}

// ---------------------------------------------------------------------------
__global__ void cvt_f32_bf16(const float* __restrict__ src,
                             bf16* __restrict__ dst, int n) {
  int i = blockIdx.x * 256 + threadIdx.x;
  if (i < n) dst[i] = (bf16)src[i];
}

// Transpose + convert: uvqk [1024][4096] f32 -> uvqk_t [4096][1024] bf16
__global__ __launch_bounds__(256) void transpose_cvt(const float* __restrict__ src,
                                                     bf16* __restrict__ dst) {
  __shared__ bf16 t[32][33];
  int r0 = blockIdx.y * 32;
  int c0 = blockIdx.x * 32;
  int r  = threadIdx.x >> 3;
  int c4 = (threadIdx.x & 7) * 4;
  for (int j = 0; j < 4; ++j)
    t[c4 + j][r] = (bf16)src[(size_t)(r0 + r) * EPROJ + c0 + c4 + j];
  __syncthreads();
  for (int j = 0; j < 4; ++j)
    dst[(size_t)(c0 + r) * DIM + r0 + c4 + j] = t[r][c4 + j];
}

// ---------------------------------------------------------------------------
// LayerNorm over D=1024, one block per (b,n) row, output bf16
// ---------------------------------------------------------------------------
__global__ __launch_bounds__(256) void ln1_kernel(const float* __restrict__ x,
                                                  const float* __restrict__ g,
                                                  const float* __restrict__ be,
                                                  bf16* __restrict__ out) {
  int row = blockIdx.x;
  int t   = threadIdx.x;
  const float* xr = x + (size_t)row * DIM;
  float v[4], s1 = 0.f, s2 = 0.f;
  for (int i = 0; i < 4; ++i) {
    float a = xr[t + i * 256];
    v[i] = a; s1 += a; s2 += a * a;
  }
  __shared__ float r1[256], r2[256];
  r1[t] = s1; r2[t] = s2;
  __syncthreads();
  for (int off = 128; off > 0; off >>= 1) {
    if (t < off) { r1[t] += r1[t + off]; r2[t] += r2[t + off]; }
    __syncthreads();
  }
  float mean = r1[0] * (1.0f / DIM);
  float var  = r2[0] * (1.0f / DIM) - mean * mean;
  float rstd = rsqrtf(var + 1e-5f);
  for (int i = 0; i < 4; ++i) {
    int c = t + i * 256;
    out[(size_t)row * DIM + c] = (bf16)((v[i] - mean) * rstd * g[c] + be[c]);
  }
}

// LayerNorm(attn) * u -> bf16.  u lives in proj[:, 0:1024] (row stride EPROJ).
__global__ __launch_bounds__(256) void ln2_kernel(const float* __restrict__ attn,
                                                  const float* __restrict__ g,
                                                  const float* __restrict__ be,
                                                  const bf16* __restrict__ proj,
                                                  bf16* __restrict__ out) {
  int row = blockIdx.x;
  int t   = threadIdx.x;
  const float* xr = attn + (size_t)row * DIM;
  const bf16*  ur = proj + (size_t)row * EPROJ;
  float v[4], s1 = 0.f, s2 = 0.f;
  for (int i = 0; i < 4; ++i) {
    float a = xr[t + i * 256];
    v[i] = a; s1 += a; s2 += a * a;
  }
  __shared__ float r1[256], r2[256];
  r1[t] = s1; r2[t] = s2;
  __syncthreads();
  for (int off = 128; off > 0; off >>= 1) {
    if (t < off) { r1[t] += r1[t + off]; r2[t] += r2[t + off]; }
    __syncthreads();
  }
  float mean = r1[0] * (1.0f / DIM);
  float var  = r2[0] * (1.0f / DIM) - mean * mean;
  float rstd = rsqrtf(var + 1e-5f);
  for (int i = 0; i < 4; ++i) {
    int c = t + i * 256;
    float ln = (v[i] - mean) * rstd * g[c] + be[c];
    out[(size_t)row * DIM + c] = (bf16)(ln * (float)ur[c]);
  }
}

// ---------------------------------------------------------------------------
// GEMM1: proj = silu(normed @ uvqk) -> bf16.  BK=64, async double-buffered
// LDS staging: tile t+1 DMAs while tile t computes.  No VGPR staging.
// ---------------------------------------------------------------------------
__global__ __launch_bounds__(256, 1) void gemm1_kernel(const bf16* __restrict__ A,
                                                       const bf16* __restrict__ Wt,
                                                       bf16* __restrict__ proj) {
  __shared__ __align__(16) bf16 As[2][128][64];
  __shared__ __align__(16) bf16 Bs[2][128][64];
  int tid = threadIdx.x, lane = tid & 31, wave = tid >> 5;
  int wm = wave >> 2, wn = wave & 3;
  int rowBase = blockIdx.y * 128;
  int colBase = blockIdx.x * 128;
  int ld_r = tid >> 1, ld_s = (tid & 1) * 32;
  const bf16* ga = A  + (size_t)(rowBase + ld_r) * DIM + ld_s;
  const bf16* gb = Wt + (size_t)(colBase + ld_r) * DIM + ld_s;

  auto stage = [&](int k0, int buf) {
    const bf16* sa = ga + k0;
    bf16* da = &As[buf][ld_r][ld_s];
    for (int j = 0; j < 4; ++j) async_copy_b128(da + 8 * j, sa + 8 * j);
    const bf16* sb = gb + k0;
    bf16* db = &Bs[buf][ld_r][ld_s];
    for (int j = 0; j < 4; ++j) async_copy_b128(db + 8 * j, sb + 8 * j);
  };

  stage(0, 0);
  wait_async0();
  __syncthreads();

  v8f acc[4][2] = {};
  for (int k0 = 0; k0 < DIM; k0 += 64) {
    int cur = (k0 >> 6) & 1;
    if (k0 + 64 < DIM) {
      stage(k0 + 64, cur ^ 1);            // overlap next tile DMA with compute
      if (k0 + 128 < DIM) {
        __builtin_prefetch(ga + k0 + 128, 0, 3);
        __builtin_prefetch(gb + k0 + 128, 0, 3);
      }
    }
    for (int kk = 0; kk < 64; kk += 32) {
      v16bf af[4], bfr[2];
      for (int mf = 0; mf < 4; ++mf)
        af[mf] = load_fragA(&As[cur][0][0], 64, wm * 64 + mf * 16, kk, lane);
      for (int nf = 0; nf < 2; ++nf)
        bfr[nf] = load_fragB(&Bs[cur][0][0], 64, wn * 32 + nf * 16, kk, lane);
      for (int mf = 0; mf < 4; ++mf)
        for (int nf = 0; nf < 2; ++nf)
          acc[mf][nf] = wmma_bf16(af[mf], bfr[nf], acc[mf][nf]);
    }
    wait_async0();     // next tile landed
    __syncthreads();   // everyone done reading current buffers
  }

  int half = lane >> 4, ln = lane & 15;
  for (int mf = 0; mf < 4; ++mf)
    for (int nf = 0; nf < 2; ++nf) {
      int col = colBase + wn * 32 + nf * 16 + ln;
      for (int r = 0; r < 8; ++r) {
        int gr = rowBase + wm * 64 + mf * 16 + r + half * 8;
        proj[(size_t)gr * EPROJ + col] = (bf16)silu_f(acc[mf][nf][r]);
      }
    }
}

// ---------------------------------------------------------------------------
// V transpose: proj v-region -> vt[(b*H+h)*128+d][n]
// ---------------------------------------------------------------------------
__global__ __launch_bounds__(256) void transpose_v(const bf16* __restrict__ proj,
                                                   bf16* __restrict__ vt) {
  __shared__ __align__(16) bf16 Ls[64][128];
  int bh = blockIdx.x;
  int nt = blockIdx.y;
  int b = bh >> 3, h = bh & 7;
  int tid = threadIdx.x;
  {
    int r = tid >> 2, seg = (tid & 3) * 32;
    const uint4* src = reinterpret_cast<const uint4*>(
        proj + (size_t)(b * SEQ + nt * 64 + r) * EPROJ + 1024 + h * 128 + seg);
    uint4* dst = reinterpret_cast<uint4*>(&Ls[r][seg]);
    dst[0] = src[0]; dst[1] = src[1]; dst[2] = src[2]; dst[3] = src[3];
  }
  __syncthreads();
  {
    int d  = tid >> 1;
    int hn = (tid & 1) * 32;
    union { bf16 h16[32]; uint4 q[4]; } u;
    for (int j = 0; j < 32; ++j) u.h16[j] = Ls[hn + j][d];
    uint4* dst = reinterpret_cast<uint4*>(
        vt + ((size_t)bh * 128 + d) * SEQ + nt * 64 + hn);
    dst[0] = u.q[0]; dst[1] = u.q[1]; dst[2] = u.q[2]; dst[3] = u.q[3];
  }
}

// ---------------------------------------------------------------------------
// Fused attention with async double-buffered K/V staging.
// S = silu(QK^T + pos_bias + ts_bias)/N, causal + key-pad masked; O = S V.
// ---------------------------------------------------------------------------
__global__ __launch_bounds__(256, 1) void attn_kernel(const bf16* __restrict__ proj,
                                                      const bf16* __restrict__ vt,
                                                      const int*  __restrict__ tstamp,
                                                      const float* __restrict__ ts_w,
                                                      const float* __restrict__ pos_w,
                                                      float* __restrict__ attn_out) {
  __shared__ __align__(16) bf16 Qs[64][128];
  __shared__ __align__(16) bf16 Ks[2][64][128];  // double-buffered
  __shared__ __align__(16) bf16 Vt[2][128][64];  // double-buffered, [d][m]
  __shared__ __align__(16) bf16 Ss[64][64];
  int bh = blockIdx.x;
  int qt = blockIdx.y;
  int b = bh >> 3, h = bh & 7;
  int tid = threadIdx.x, lane = tid & 31, wave = tid >> 5;
  int half = lane >> 4, ln = lane & 15;
  int wm = wave >> 1, wn = wave & 1;
  int len_b = SEQ - 100 * b;
  const int* tsb = tstamp + b * SEQ;
  const size_t qoff = 2048 + h * 128;
  const size_t koff = 3072 + h * 128;

  int kv_r  = tid >> 2, kv_seg = (tid & 3) * 32;   // K staging: 32 bf16/thread
  int v_d   = tid >> 1, v_hn   = (tid & 1) * 32;   // V staging: 32 bf16/thread

  auto stage_kv = [&](int kc, int buf) {
    const bf16* ksrc = proj + (size_t)(b * SEQ + kc * 64 + kv_r) * EPROJ + koff + kv_seg;
    bf16* kdst = &Ks[buf][kv_r][kv_seg];
    for (int j = 0; j < 4; ++j) async_copy_b128(kdst + 8 * j, ksrc + 8 * j);
    const bf16* vsrc = vt + ((size_t)bh * 128 + v_d) * SEQ + kc * 64 + v_hn;
    bf16* vdst = &Vt[buf][v_d][v_hn];
    for (int j = 0; j < 4; ++j) async_copy_b128(vdst + 8 * j, vsrc + 8 * j);
  };

  // stage chunk 0 + Q tile asynchronously
  stage_kv(0, 0);
  {
    const bf16* qsrc = proj + (size_t)(b * SEQ + qt * 64 + kv_r) * EPROJ + qoff + kv_seg;
    bf16* qdst = &Qs[kv_r][kv_seg];
    for (int j = 0; j < 4; ++j) async_copy_b128(qdst + 8 * j, qsrc + 8 * j);
  }

  // hoist per-row bias state out of the key-chunk loop
  int gi_row[8], ts_row[8];
  const float* posw_row[8];
  for (int r = 0; r < 8; ++r) {
    int gi = qt * 64 + wm * 16 + r + half * 8;
    gi_row[r] = gi;
    ts_row[r] = tsb[imin(gi + 1, SEQ - 1)];
    posw_row[r] = pos_w + (SEQ - 1 - gi);
  }

  wait_async0();
  __syncthreads();

  v8f oacc[4] = {};

  for (int kc = 0; kc <= qt; ++kc) {
    int cur = kc & 1;
    if (kc < qt) stage_kv(kc + 1, cur ^ 1);   // overlap next chunk DMA

    // ---- S = Q K^T
    v8f sacc[2] = {};
    for (int d0 = 0; d0 < 128; d0 += 32) {
      v16bf aq = load_fragA(&Qs[0][0], 128, wm * 16, d0, lane);
      for (int nf = 0; nf < 2; ++nf) {
        v16bf bk = load_fragB(&Ks[cur][0][0], 128, wn * 32 + nf * 16, d0, lane);
        sacc[nf] = wmma_bf16(aq, bk, sacc[nf]);
      }
    }
    // ---- bias + silu + masks -> bf16 S in LDS
    for (int nf = 0; nf < 2; ++nf) {
      int lm = wn * 32 + nf * 16 + ln;
      int gm = kc * 64 + lm;
      int ts_col = tsb[gm];
      bool key_pad = (gm >= len_b);
      for (int r = 0; r < 8; ++r) {
        float s = sacc[nf][r];
        float ad = fmaxf(fabsf((float)(ts_row[r] - ts_col)), 1.0f);
        int bucket = imax(0, imin((int)(__logf(ad) * (1.0f / 0.301f)), NBUCK));
        s += posw_row[r][gm] + ts_w[bucket];
        s = silu_f(s) * (1.0f / SEQ);
        if (gm > gi_row[r] || key_pad) s = 0.0f;
        Ss[wm * 16 + r + half * 8][lm] = (bf16)s;
      }
    }
    __syncthreads();

    // ---- O += S @ V
    for (int kk = 0; kk < 64; kk += 32) {
      v16bf bv = load_fragB(&Vt[cur][0][0], 64, wave * 16, kk, lane);
      for (int mf = 0; mf < 4; ++mf) {
        v16bf as = load_fragA(&Ss[0][0], 64, mf * 16, kk, lane);
        oacc[mf] = wmma_bf16(as, bv, oacc[mf]);
      }
    }
    wait_async0();     // next chunk landed
    __syncthreads();   // everyone done with Ss / current buffers
  }

  int col = wave * 16 + ln;
  for (int mf = 0; mf < 4; ++mf)
    for (int r = 0; r < 8; ++r) {
      int gi = qt * 64 + mf * 16 + r + half * 8;
      attn_out[((size_t)(b * SEQ + gi)) * DIM + h * 128 + col] = oacc[mf][r];
    }
}

// ---------------------------------------------------------------------------
// GEMM2: out = g @ o_w^T + o_b + x, pad-masked.  Same async pipeline as GEMM1.
// ---------------------------------------------------------------------------
__global__ __launch_bounds__(256, 1) void gemm2_kernel(const bf16* __restrict__ G,
                                                       const bf16* __restrict__ Ow,
                                                       const float* __restrict__ o_b,
                                                       const float* __restrict__ x,
                                                       float* __restrict__ out) {
  __shared__ __align__(16) bf16 As[2][128][64];
  __shared__ __align__(16) bf16 Bs[2][128][64];
  int tid = threadIdx.x, lane = tid & 31, wave = tid >> 5;
  int wm = wave >> 2, wn = wave & 3;
  int rowBase = blockIdx.y * 128;
  int colBase = blockIdx.x * 128;
  int ld_r = tid >> 1, ld_s = (tid & 1) * 32;
  const bf16* ga = G  + (size_t)(rowBase + ld_r) * DIM + ld_s;
  const bf16* gb = Ow + (size_t)(colBase + ld_r) * DIM + ld_s;

  auto stage = [&](int k0, int buf) {
    const bf16* sa = ga + k0;
    bf16* da = &As[buf][ld_r][ld_s];
    for (int j = 0; j < 4; ++j) async_copy_b128(da + 8 * j, sa + 8 * j);
    const bf16* sb = gb + k0;
    bf16* db = &Bs[buf][ld_r][ld_s];
    for (int j = 0; j < 4; ++j) async_copy_b128(db + 8 * j, sb + 8 * j);
  };

  stage(0, 0);
  wait_async0();
  __syncthreads();

  v8f acc[4][2] = {};
  for (int k0 = 0; k0 < DIM; k0 += 64) {
    int cur = (k0 >> 6) & 1;
    if (k0 + 64 < DIM) {
      stage(k0 + 64, cur ^ 1);
      if (k0 + 128 < DIM) {
        __builtin_prefetch(ga + k0 + 128, 0, 3);
        __builtin_prefetch(gb + k0 + 128, 0, 3);
      }
    }
    for (int kk = 0; kk < 64; kk += 32) {
      v16bf af[4], bfr[2];
      for (int mf = 0; mf < 4; ++mf)
        af[mf] = load_fragA(&As[cur][0][0], 64, wm * 64 + mf * 16, kk, lane);
      for (int nf = 0; nf < 2; ++nf)
        bfr[nf] = load_fragB(&Bs[cur][0][0], 64, wn * 32 + nf * 16, kk, lane);
      for (int mf = 0; mf < 4; ++mf)
        for (int nf = 0; nf < 2; ++nf)
          acc[mf][nf] = wmma_bf16(af[mf], bfr[nf], acc[mf][nf]);
    }
    wait_async0();
    __syncthreads();
  }

  int half = lane >> 4, ln = lane & 15;
  for (int mf = 0; mf < 4; ++mf)
    for (int nf = 0; nf < 2; ++nf) {
      int col = colBase + wn * 32 + nf * 16 + ln;
      for (int r = 0; r < 8; ++r) {
        int gr = rowBase + wm * 64 + mf * 16 + r + half * 8;
        int b = gr >> 11, n = gr & (SEQ - 1);
        float val = acc[mf][nf][r] + o_b[col] + x[(size_t)gr * DIM + col];
        if (n >= SEQ - 100 * b) val = 0.0f;
        out[(size_t)gr * DIM + col] = val;
      }
    }
}

// ---------------------------------------------------------------------------
extern "C" void kernel_launch(void* const* d_in, const int* in_sizes, int n_in,
                              void* d_out, int out_size, void* d_ws, size_t ws_size,
                              hipStream_t stream) {
  (void)in_sizes; (void)n_in; (void)out_size; (void)ws_size;
  const float* x      = (const float*)d_in[0];
  const int*   tstamp = (const int*)d_in[1];
  const float* uvqk   = (const float*)d_in[4];
  const float* o_w    = (const float*)d_in[5];
  const float* o_b    = (const float*)d_in[6];
  const float* nx_g   = (const float*)d_in[7];
  const float* nx_b   = (const float*)d_in[8];
  const float* na_g   = (const float*)d_in[9];
  const float* na_b   = (const float*)d_in[10];
  const float* ts_w   = (const float*)d_in[11];
  const float* pos_w  = (const float*)d_in[12];
  float* out = (float*)d_out;

  const size_t BN = (size_t)BATCH * SEQ;
  char* ws = (char*)d_ws;
  size_t off = 0;
  bf16*  normed = (bf16*)(ws + off);  off += BN * DIM * sizeof(bf16);
  bf16*  uvqk_t = (bf16*)(ws + off);  off += (size_t)DIM * EPROJ * sizeof(bf16);
  bf16*  ow_b   = (bf16*)(ws + off);  off += (size_t)DIM * DIM * sizeof(bf16);
  bf16*  proj   = (bf16*)(ws + off);  off += BN * EPROJ * sizeof(bf16);
  bf16*  vt     = (bf16*)(ws + off);  off += BN * DIM * sizeof(bf16);
  float* att_ws = (float*)(ws + off); off += BN * DIM * sizeof(float);
  bf16*  g_ws   = (bf16*)(ws + off);  off += BN * DIM * sizeof(bf16);

  transpose_cvt<<<dim3(EPROJ / 32, DIM / 32), 256, 0, stream>>>(uvqk, uvqk_t);
  cvt_f32_bf16<<<(DIM * DIM + 255) / 256, 256, 0, stream>>>(o_w, ow_b, DIM * DIM);
  ln1_kernel<<<(int)BN, 256, 0, stream>>>(x, nx_g, nx_b, normed);
  gemm1_kernel<<<dim3(EPROJ / 128, (int)(BN / 128)), 256, 0, stream>>>(
      normed, uvqk_t, proj);
  transpose_v<<<dim3(BATCH * HEADS, SEQ / 64), 256, 0, stream>>>(proj, vt);
  attn_kernel<<<dim3(BATCH * HEADS, SEQ / 64), 256, 0, stream>>>(
      proj, vt, tstamp, ts_w, pos_w, att_ws);
  ln2_kernel<<<(int)BN, 256, 0, stream>>>(att_ws, na_g, na_b, proj, g_ws);
  gemm2_kernel<<<dim3(DIM / 128, (int)(BN / 128)), 256, 0, stream>>>(
      g_ws, ow_b, o_b, x, out);
}
